// GraphAttentionLayer_56212531970311
// MI455X (gfx1250) — matrix operand order repaired and checked by reading, compiled, not run on hardware
//
#include <hip/hip_runtime.h>
#include <math.h>

// GraphAttentionLayer for MI455X (gfx1250, wave32, WMMA), flash-style 2-pass.
// A-tiles staged in LDS; dual independent WMMA accumulator chains for ILP;
// single-exp online softmax update.
#define NN 8192
#define DD 128
#define LN_EPS 1e-5f
#define INV_SQRT_D 0.08838834764831845f  // 1/sqrt(128)
#define LDSW 136                          // padded LDS row stride (halves)

typedef _Float16 half8 __attribute__((ext_vector_type(8)));
typedef _Float16 v16h  __attribute__((ext_vector_type(16)));
typedef float    v8f   __attribute__((ext_vector_type(8)));

union AFrag { v16h v; half8 h[2]; };
union CFrag { v8f  v; float f[8]; };

// ---------------------------------------------------------------- utilities

__global__ void k_zero_u32(unsigned* __restrict__ p, int nw) {
  int i = blockIdx.x * blockDim.x + threadIdx.x;
  if (i < nw) p[i] = 0u;
}

__global__ void k_scatter_edges(const int* __restrict__ ei, int E,
                                unsigned* __restrict__ bm) {
  int e = blockIdx.x * blockDim.x + threadIdx.x;
  if (e >= E) return;
  int r = ei[e];        // edge_index[0, e]
  int c = ei[E + e];    // edge_index[1, e]
  atomicOr(&bm[r * (NN / 32) + (c >> 5)], 1u << (c & 31));
}

// ------------------------------------------------- pos_emb: [N,3]x[128,3]^T
__global__ void k_posemb(const float* __restrict__ pos,
                         const float* __restrict__ Wp,
                         const float* __restrict__ bp,
                         _Float16* __restrict__ P) {
  int n = blockIdx.x, d = threadIdx.x;
  __shared__ float ps[3];
  __shared__ float red[DD];
  if (d < 3) ps[d] = pos[n * 3 + d];
  __syncthreads();
  float v = bp[d] + ps[0] * Wp[d * 3 + 0] + ps[1] * Wp[d * 3 + 1]
                  + ps[2] * Wp[d * 3 + 2];
  red[d] = v * v;
  __syncthreads();
  for (int off = 64; off > 0; off >>= 1) {
    if (d < off) red[d] += red[d + off];
    __syncthreads();
  }
  float dn = fmaxf(sqrtf(red[0]), 1e-12f);
  P[(size_t)n * DD + d] = (_Float16)(v / dn);
}

// ------------------------------------- Q/K/V projection + LayerNorm (f32)
__global__ void k_qkv(const float* __restrict__ x,
                      const float* __restrict__ Wq, const float* __restrict__ bq,
                      const float* __restrict__ Wk, const float* __restrict__ bk,
                      const float* __restrict__ Wv, const float* __restrict__ bv,
                      const float* __restrict__ gq, const float* __restrict__ oq,
                      const float* __restrict__ gk, const float* __restrict__ ok,
                      const float* __restrict__ gv, const float* __restrict__ ov,
                      _Float16* __restrict__ Qh, _Float16* __restrict__ Kh,
                      _Float16* __restrict__ VT) {
  int n = blockIdx.x, which = blockIdx.y, d = threadIdx.x;
  const float *W, *b, *g, *o;
  if (which == 0)      { W = Wq; b = bq; g = gq; o = oq; }
  else if (which == 1) { W = Wk; b = bk; g = gk; o = ok; }
  else                 { W = Wv; b = bv; g = gv; o = ov; }

  __shared__ float xs[DD];
  __shared__ float red[DD];
  xs[d] = x[(size_t)n * DD + d];
  __syncthreads();

  const float* wr = W + (size_t)d * DD;
  float acc = b[d];
#pragma unroll 8
  for (int i = 0; i < DD; i++) acc += xs[i] * wr[i];

  red[d] = acc;
  __syncthreads();
  for (int off = 64; off > 0; off >>= 1) {
    if (d < off) red[d] += red[d + off];
    __syncthreads();
  }
  float mean = red[0] * (1.0f / DD);
  __syncthreads();
  float dv = acc - mean;
  red[d] = dv * dv;
  __syncthreads();
  for (int off = 64; off > 0; off >>= 1) {
    if (d < off) red[d] += red[d + off];
    __syncthreads();
  }
  float var = red[0] * (1.0f / DD);
  float y = dv * rsqrtf(var + LN_EPS) * g[d] + o[d];

  if (which == 0)      Qh[(size_t)n * DD + d] = (_Float16)(y * INV_SQRT_D);
  else if (which == 1) Kh[(size_t)n * DD + d] = (_Float16)y;
  else                 VT[(size_t)d * NN + n] = (_Float16)y;  // transposed for WMMA B
}

// ---- one 16x16 score tile: (Q/sqrtD)@K^T + P@P^T, f32 accumulate.
// A rows from LDS (conflict-free b128, hoisted to regs by the compiler),
// B rows from global (L2-resident). Two independent accumulator chains
// (QK and PP) double matrix-pipe ILP and let both load pairs clause.
__device__ __forceinline__ v8f score_tile16(const _Float16* sq,
                                            const _Float16* sp,
                                            const _Float16* __restrict__ kb,
                                            const _Float16* __restrict__ pb,
                                            int l, int hb) {
  const _Float16* qrow = sq + l * LDSW;
  const _Float16* prow = sp + l * LDSW;
  v8f cq = {}, cp = {};
#pragma unroll
  for (int kc = 0; kc < DD; kc += 32) {
    v16h bk_ = *(const v16h*)(kb + kc + hb * 16);
    v16h bp_ = *(const v16h*)(pb + kc + hb * 16);
    AFrag aq, ap;
    aq.h[0] = *(const half8*)(qrow + kc + hb * 16);
    aq.h[1] = *(const half8*)(qrow + kc + hb * 16 + 8);
    ap.h[0] = *(const half8*)(prow + kc + hb * 16);
    ap.h[1] = *(const half8*)(prow + kc + hb * 16 + 8);
    cq = __builtin_amdgcn_wmma_f32_16x16x32_f16(false, aq.v, false, bk_,
                                                (short)0, cq, false, false);
    cp = __builtin_amdgcn_wmma_f32_16x16x32_f16(false, ap.v, false, bp_,
                                                (short)0, cp, false, false);
  }
  return cq + cp;
}

// stage the block's 16 Q rows and 16 P rows into padded LDS tiles (coalesced)
__device__ __forceinline__ void stage_a_tiles(const _Float16* __restrict__ Qh,
                                              const _Float16* __restrict__ P,
                                              _Float16* sq, _Float16* sp,
                                              int m0, int tid) {
  int srow = tid >> 4;
  int scol = (tid & 15) * 8;
  *(half8*)(sq + srow * LDSW + scol) =
      *(const half8*)(Qh + (size_t)(m0 + srow) * DD + scol);
  *(half8*)(sp + srow * LDSW + scol) =
      *(const half8*)(P + (size_t)(m0 + srow) * DD + scol);
}

// ---- pass 1: per-row softmax stats (max, sum-exp) via WMMA score recompute.
__global__ __launch_bounds__(256)
void k_stats(const _Float16* __restrict__ Qh,
             const _Float16* __restrict__ Kh,
             const _Float16* __restrict__ P,
             const unsigned* __restrict__ bm,
             float* __restrict__ rmax, float* __restrict__ rsum) {
  int m0 = blockIdx.x * 16;
  int tid = threadIdx.x;
  int lane = tid & 31, w = tid >> 5;
  int l = lane & 15, hb = lane >> 4;

  __shared__ __align__(32) _Float16 sq[16 * LDSW];
  __shared__ __align__(32) _Float16 sp[16 * LDSW];
  stage_a_tiles(Qh, P, sq, sp, m0, tid);
  __syncthreads();

  float mr[8], sr[8];
#pragma unroll
  for (int r = 0; r < 8; r++) { mr[r] = -3.0e38f; sr[r] = 0.0f; }

  for (int n0 = w * 16; n0 < NN; n0 += 128) {
    const _Float16* kb = Kh + (size_t)(n0 + l) * DD;
    const _Float16* pb = P  + (size_t)(n0 + l) * DD;
    CFrag cf; cf.v = score_tile16(sq, sp, kb, pb, l, hb);
    int gn = n0 + l;
#pragma unroll
    for (int r = 0; r < 8; r++) {
      int gm = m0 + r + hb * 8;
      unsigned wd = bm[gm * (NN / 32) + (gn >> 5)];
      float v = ((wd >> (gn & 31)) & 1u) ? cf.f[r] : -10000.0f;
      // single-exp online update (values never NaN)
      bool gt = v > mr[r];
      float e = __expf(gt ? mr[r] - v : v - mr[r]);
      sr[r] = gt ? sr[r] * e + 1.0f : sr[r] + e;
      mr[r] = gt ? v : mr[r];
    }
  }
  // merge across the 16 lanes of each half-wave (C layout: lane = column)
#pragma unroll
  for (int off = 8; off >= 1; off >>= 1) {
#pragma unroll
    for (int r = 0; r < 8; r++) {
      float m2 = __shfl_xor(mr[r], off, 32);
      float s2 = __shfl_xor(sr[r], off, 32);
      float nm = fmaxf(mr[r], m2);
      sr[r] = sr[r] * __expf(mr[r] - nm) + s2 * __expf(m2 - nm);
      mr[r] = nm;
    }
  }
  __shared__ float smax[8][16], ssum[8][16];
  if (l == 0) {
#pragma unroll
    for (int r = 0; r < 8; r++) {
      smax[w][r + hb * 8] = mr[r];
      ssum[w][r + hb * 8] = sr[r];
    }
  }
  __syncthreads();
  if (tid < 16) {
    float m = smax[0][tid], s = ssum[0][tid];
#pragma unroll
    for (int wv = 1; wv < 8; wv++) {
      float m2 = smax[wv][tid], s2 = ssum[wv][tid];
      float nm = fmaxf(m, m2);
      s = s * __expf(m - nm) + s2 * __expf(m2 - nm);
      m = nm;
    }
    rmax[m0 + tid] = m;
    rsum[m0 + tid] = s;
  }
}

// ---- pass 2: recompute scores, normalize, write final attn, and attn @ V.
__global__ __launch_bounds__(256)
void k_fused(const _Float16* __restrict__ Qh,
             const _Float16* __restrict__ Kh,
             const _Float16* __restrict__ P,
             const _Float16* __restrict__ VT,
             const unsigned* __restrict__ bm,
             const float* __restrict__ rmax,
             const float* __restrict__ rsum,
             float* __restrict__ attn, float* __restrict__ out) {
  int m0 = blockIdx.x * 16;
  int tid = threadIdx.x;
  int lane = tid & 31, w = tid >> 5;
  int l = lane & 15, hb = lane >> 4;
  int d0 = w * 16;

  __shared__ __align__(32) _Float16 sq[16 * LDSW];
  __shared__ __align__(32) _Float16 sp[16 * LDSW];
  __shared__ __align__(32) _Float16 aT[16 * LDSW];
  stage_a_tiles(Qh, P, sq, sp, m0, tid);

  const _Float16* vt = VT + (size_t)(d0 + l) * NN;

  float rmx8[8], rin8[8];
#pragma unroll
  for (int r = 0; r < 8; r++) {
    int gm = m0 + r + hb * 8;
    rmx8[r] = rmax[gm];
    rin8[r] = 1.0f / rsum[gm];
  }
  __syncthreads();

  v8f acc0 = {}, acc1 = {};   // two independent attn@V chains (kk 0,2 / 1,3)
  for (int j0 = 0; j0 < NN; j0 += 128) {
    int n0 = j0 + w * 16;
    const _Float16* kb = Kh + (size_t)(n0 + l) * DD;
    const _Float16* pb = P  + (size_t)(n0 + l) * DD;
    CFrag cf; cf.v = score_tile16(sq, sp, kb, pb, l, hb);
    int gn = n0 + l;
#pragma unroll
    for (int r = 0; r < 8; r++) {
      int gm = m0 + r + hb * 8;
      unsigned wd = bm[gm * (NN / 32) + (gn >> 5)];
      float s = ((wd >> (gn & 31)) & 1u) ? cf.f[r] : -10000.0f;
      float p = __expf(s - rmx8[r]) * rin8[r];
      attn[(size_t)gm * NN + gn] = p;                 // final attn value
      aT[(r + hb * 8) * LDSW + (w * 16 + l)] = (_Float16)p;
    }
    __syncthreads();
#pragma unroll
    for (int kk = 0; kk < 4; kk++) {
      const _Float16* ar = &aT[l * LDSW + kk * 32];
      AFrag a; a.h[0] = *(const half8*)(ar + hb * 16);
               a.h[1] = *(const half8*)(ar + hb * 16 + 8);
      v16h b = *(const v16h*)(vt + j0 + kk * 32 + hb * 16);
      if (kk & 1)
        acc1 = __builtin_amdgcn_wmma_f32_16x16x32_f16(false, a.v, false, b,
                                                      (short)0, acc1, false, false);
      else
        acc0 = __builtin_amdgcn_wmma_f32_16x16x32_f16(false, a.v, false, b,
                                                      (short)0, acc0, false, false);
    }
    __syncthreads();
  }

  CFrag cf; cf.v = acc0 + acc1;
#pragma unroll
  for (int r = 0; r < 8; r++)
    out[(size_t)(m0 + r + hb * 8) * DD + d0 + l] = cf.f[r];
}

// ----------------------------------------------------------------- launch
extern "C" void kernel_launch(void* const* d_in, const int* in_sizes, int n_in,
                              void* d_out, int out_size, void* d_ws, size_t ws_size,
                              hipStream_t stream) {
  const float* x    = (const float*)d_in[0];
  const float* pos  = (const float*)d_in[1];
  const int*   ei   = (const int*)d_in[2];
  const float* Wq   = (const float*)d_in[3];
  const float* bq   = (const float*)d_in[4];
  const float* Wk   = (const float*)d_in[5];
  const float* bk   = (const float*)d_in[6];
  const float* Wv   = (const float*)d_in[7];
  const float* bv   = (const float*)d_in[8];
  const float* Wp   = (const float*)d_in[9];
  const float* bp   = (const float*)d_in[10];
  const float* gq   = (const float*)d_in[11];
  const float* oq   = (const float*)d_in[12];
  const float* gk   = (const float*)d_in[13];
  const float* ok   = (const float*)d_in[14];
  const float* gv   = (const float*)d_in[15];
  const float* ov   = (const float*)d_in[16];
  int E = in_sizes[2] / 2;

  // workspace layout (all offsets 32B-aligned)
  char* ws = (char*)d_ws;
  unsigned* bm  = (unsigned*)ws;                                   // 8 MB bitmap
  _Float16* P   = (_Float16*)(ws + (size_t)8 * 1024 * 1024);       // 2 MB
  _Float16* Qh  = (_Float16*)(ws + (size_t)10 * 1024 * 1024);      // 2 MB
  _Float16* Kh  = (_Float16*)(ws + (size_t)12 * 1024 * 1024);      // 2 MB
  _Float16* VT  = (_Float16*)(ws + (size_t)14 * 1024 * 1024);      // 2 MB
  float*    rmx = (float*)   (ws + (size_t)16 * 1024 * 1024);      // 32 KB
  float*    rsm = (float*)   (ws + (size_t)16 * 1024 * 1024 + 32768);

  float* out  = (float*)d_out;                 // [N, D]
  float* attn = out + (size_t)NN * DD;         // [N, N]

  int nwords = NN * (NN / 32);
  k_zero_u32<<<(nwords + 255) / 256, 256, 0, stream>>>(bm, nwords);
  k_scatter_edges<<<(E + 255) / 256, 256, 0, stream>>>(ei, E, bm);
  k_posemb<<<NN, DD, 0, stream>>>(pos, Wp, bp, P);
  k_qkv<<<dim3(NN, 3), DD, 0, stream>>>(x, Wq, bq, Wk, bk, Wv, bv,
                                        gq, oq, gk, ok, gv, ov, Qh, Kh, VT);
  k_stats<<<NN / 16, 256, 0, stream>>>(Qh, Kh, P, bm, rmx, rsm);
  k_fused<<<NN / 16, 256, 0, stream>>>(Qh, Kh, P, VT, bm, rmx, rsm, attn, out);
}
